// Net_MP_68805376082316
// MI455X (gfx1250) — matrix operand order repaired and checked by reading, not compiled, not run on hardware
//
#include <hip/hip_runtime.h>

typedef __attribute__((ext_vector_type(16))) _Float16 v16h;
typedef __attribute__((ext_vector_type(8)))  float    v8f;

#define N_NODES 50000
#define N_EDGES 200000

__device__ __forceinline__ float frelu(float v) { return v > 0.f ? v : 0.f; }

// ---------------- Layer 1: per-edge NNConv(2->32) messages, scatter-add ----
__global__ void conv1_edge_kernel(const float* __restrict__ x,
                                  const long long* __restrict__ esrc,
                                  const long long* __restrict__ edst,
                                  const float* __restrict__ ea,
                                  const float* __restrict__ w1,
                                  const float* __restrict__ b1,
                                  const float* __restrict__ w2,
                                  const float* __restrict__ b2,
                                  float* __restrict__ agg1)
{
    __shared__ float sw2[16 * 64];
    __shared__ float sw1[32];
    __shared__ float sb1[16];
    __shared__ float sb2[64];
    for (int i = threadIdx.x; i < 16 * 64; i += blockDim.x) sw2[i] = w2[i];
    if (threadIdx.x < 32) sw1[threadIdx.x] = w1[threadIdx.x];
    if (threadIdx.x < 16) sb1[threadIdx.x] = b1[threadIdx.x];
    if (threadIdx.x < 64) sb2[threadIdx.x] = b2[threadIdx.x];
    __syncthreads();

    int e = blockIdx.x * blockDim.x + threadIdx.x;
    if (e >= N_EDGES) return;

    float ea0 = ea[2 * e], ea1 = ea[2 * e + 1];
    float r[16];
#pragma unroll
    for (int k = 0; k < 16; ++k)
        r[k] = frelu(ea0 * sw1[k] + ea1 * sw1[16 + k] + sb1[k]);

    int s = (int)esrc[e];
    int d = (int)edst[e];
    float x0 = x[2 * s], x1 = x[2 * s + 1];
    float* dstp = agg1 + (long)d * 32;
#pragma unroll 4
    for (int o = 0; o < 32; ++o) {
        float a0 = sb2[o], a1 = sb2[32 + o];
#pragma unroll
        for (int k = 0; k < 16; ++k) {
            a0 += r[k] * sw2[k * 64 + o];
            a1 += r[k] * sw2[k * 64 + 32 + o];
        }
        atomicAdd(dstp + o, x0 * a0 + x1 * a1);
    }
}

// ---------------- Layer 1: node update h1 = relu(agg + x@root1 + b) (in place)
__global__ void node1_kernel(const float* __restrict__ x,
                             const float* __restrict__ root1,
                             const float* __restrict__ bias1,
                             float* __restrict__ buf0)
{
    int idx = blockIdx.x * blockDim.x + threadIdx.x;
    if (idx >= N_NODES * 32) return;
    int n = idx >> 5, o = idx & 31;
    float v = buf0[idx] + x[2 * n] * root1[o] + x[2 * n + 1] * root1[32 + o] + bias1[o];
    buf0[idx] = frelu(v);
}

// ---------------- Layer 2: fused edge-MLP + per-edge matvec as WMMA GEMM ----
// msg[E,32] = P[E,544] @ T'[544,32],  P[e, k*32+i] = r2[e,k]*h1[src,i] (k<16),
// P[e,512+i] = h1[src,i];  T'[k*32+i, o] = w2[k, i*32+o];  T'[512+i, o] = b2[i*32+o].
// One wave per 16-edge tile; 17 K-steps of v_wmma_f32_16x16x32_f16 x 2 N-tiles.
// T' is pre-swizzled in LDS into per-lane WMMA B fragments: fragment f = t*2+c
// holds, for lane L, the v16h with halves j = T'[32t + (L>=16?8:0) + j + (j>=8?8:0)][16c + (L&15)],
// so the inner loop fetches each B fragment with two ds_load_b128.
__global__ void __launch_bounds__(256) conv2_edge_wmma_kernel(
    const float* __restrict__ h1,
    const long long* __restrict__ esrc,
    const long long* __restrict__ edst,
    const float* __restrict__ ea,
    const float* __restrict__ w1,
    const float* __restrict__ b1,
    const float* __restrict__ w2,
    const float* __restrict__ b2,
    float* __restrict__ agg2)
{
    __shared__ v16h  TmFrag[34 * 32];   // 34 KB: swizzled B fragments (f16)
    __shared__ float sw1[32];
    __shared__ float sb1[16];

    {
        _Float16* tmh = (_Float16*)TmFrag;
        for (int i = threadIdx.x; i < 34 * 32 * 16; i += blockDim.x) {
            int f    = i >> 9;          // fragment: t*2 + c
            int r    = i & 511;
            int ln   = r >> 4;          // lane within fragment
            int j    = r & 15;          // half index within v16h
            int t    = f >> 1;
            int c    = f & 1;
            int K    = (t << 5) + ((ln >= 16) ? 8 : 0) + j + ((j >= 8) ? 8 : 0);
            int col  = (c << 4) + (ln & 15);
            float v  = (K < 512) ? w2[(K >> 5) * 1024 + (K & 31) * 32 + col]
                                 : b2[(K - 512) * 32 + col];
            tmh[i] = (_Float16)v;
        }
    }
    if (threadIdx.x < 32) sw1[threadIdx.x] = w1[threadIdx.x];
    if (threadIdx.x < 16) sb1[threadIdx.x] = b1[threadIdx.x];
    __syncthreads();

    const int lane  = threadIdx.x & 31;          // wave32 (gfx1250)
    const int wave  = threadIdx.x >> 5;
    const int gwave = blockIdx.x * (blockDim.x >> 5) + wave;
    const int nwave = gridDim.x * (blockDim.x >> 5);
    const int NT    = N_EDGES / 16;              // 12500 tiles, exact

    const int el  = lane & 15;                   // this lane's edge within tile
    const int hiA = (lane >> 4) << 3;            // 0 / 8 : A K-group & D row base

    for (int tile = gwave; tile < NT; tile += nwave) {
        const int e = tile * 16 + el;
        const float ea0 = ea[2 * e], ea1 = ea[2 * e + 1];
        float rr[17];
#pragma unroll
        for (int t = 0; t < 16; ++t)
            rr[t] = frelu(ea0 * sw1[t] + ea1 * sw1[16 + t] + sb1[t]);
        rr[16] = 1.0f;                            // bias row coefficient

        const int s = (int)esrc[e];
        v16h xsh;                                 // the 16 fixed feature offs, f16
#pragma unroll
        for (int j = 0; j < 16; ++j) {
            int off = hiA + j + ((j >= 8) ? 8 : 0);   // A-layout K offsets
            xsh[j] = (_Float16)h1[s * 32 + off];
        }

        v8f c0 = {};                              // D cols 0..15
        v8f c1 = {};                              // D cols 16..31
#pragma unroll 2
        for (int t = 0; t < 17; ++t) {
            const v16h a  = xsh * (_Float16)rr[t];        // v_pk_mul_f16 x8
            const v16h b0 = TmFrag[(t * 2 + 0) * 32 + lane];  // 2x ds_load_b128
            const v16h b1 = TmFrag[(t * 2 + 1) * 32 + lane];
            c0 = __builtin_amdgcn_wmma_f32_16x16x32_f16(
                     false, a, false, b0, (short)0, c0, false, false);
            c1 = __builtin_amdgcn_wmma_f32_16x16x32_f16(
                     false, a, false, b1, (short)0, c1, false, false);
        }

        // Scatter D (16 edges x 32 outs) with f32 atomics. Lane holds rows
        // hiA..hiA+7 at column el (c0) and el+16 (c1).
#pragma unroll
        for (int v = 0; v < 8; ++v) {
            const int m = hiA + v;
            const int d = (int)edst[tile * 16 + m];
            atomicAdd(agg2 + (long)d * 32 + el,      c0[v]);
            atomicAdd(agg2 + (long)d * 32 + 16 + el, c1[v]);
        }
    }
}

// ---------------- Layer 2 node update + fc2 head -----------------------------
__global__ void node2_kernel(const float* __restrict__ h1,
                             const float* __restrict__ agg2,
                             const float* __restrict__ root2,
                             const float* __restrict__ bias2,
                             const float* __restrict__ fcw,
                             const float* __restrict__ fcb,
                             float* __restrict__ out)
{
    __shared__ float sr[32 * 32];
    __shared__ float sw[32];
    __shared__ float sb[32];
    for (int i = threadIdx.x; i < 32 * 32; i += blockDim.x) sr[i] = root2[i];
    if (threadIdx.x < 32) {
        sw[threadIdx.x] = fcw[threadIdx.x];
        sb[threadIdx.x] = bias2[threadIdx.x];
    }
    __syncthreads();

    int n = blockIdx.x * blockDim.x + threadIdx.x;
    if (n >= N_NODES) return;

    float hx[32];
#pragma unroll
    for (int i = 0; i < 32; ++i) hx[i] = h1[n * 32 + i];

    float acc = fcb[0];
#pragma unroll 4
    for (int o = 0; o < 32; ++o) {
        float v = agg2[n * 32 + o] + sb[o];
#pragma unroll
        for (int i = 0; i < 32; ++i) v += hx[i] * sr[i * 32 + o];
        acc += frelu(v) * sw[o];
    }
    out[n] = acc;
}

// ---------------------------------------------------------------------------
extern "C" void kernel_launch(void* const* d_in, const int* in_sizes, int n_in,
                              void* d_out, int out_size, void* d_ws, size_t ws_size,
                              hipStream_t stream)
{
    const float*     x      = (const float*)d_in[0];
    const long long* eidx   = (const long long*)d_in[1];   // jnp.int64, [2,E]
    const float*     ea     = (const float*)d_in[2];
    const float*     nn1_w1 = (const float*)d_in[3];
    const float*     nn1_b1 = (const float*)d_in[4];
    const float*     nn1_w2 = (const float*)d_in[5];
    const float*     nn1_b2 = (const float*)d_in[6];
    const float*     root1  = (const float*)d_in[7];
    const float*     bias1  = (const float*)d_in[8];
    const float*     nn2_w1 = (const float*)d_in[9];
    const float*     nn2_b1 = (const float*)d_in[10];
    const float*     nn2_w2 = (const float*)d_in[11];
    const float*     nn2_b2 = (const float*)d_in[12];
    const float*     root2  = (const float*)d_in[13];
    const float*     bias2  = (const float*)d_in[14];
    const float*     fc2_w  = (const float*)d_in[15];
    const float*     fc2_b  = (const float*)d_in[16];

    const long long* esrc = eidx;
    const long long* edst = eidx + N_EDGES;

    float* buf0 = (float*)d_ws;                 // agg1, then h1 in place
    float* buf1 = buf0 + (size_t)N_NODES * 32;  // agg2
    hipMemsetAsync(d_ws, 0, (size_t)2 * N_NODES * 32 * sizeof(float), stream);

    conv1_edge_kernel<<<(N_EDGES + 255) / 256, 256, 0, stream>>>(
        x, esrc, edst, ea, nn1_w1, nn1_b1, nn1_w2, nn1_b2, buf0);

    node1_kernel<<<(N_NODES * 32 + 255) / 256, 256, 0, stream>>>(
        x, root1, bias1, buf0);

    conv2_edge_wmma_kernel<<<128, 256, 0, stream>>>(
        buf0, esrc, edst, ea, nn2_w1, nn2_b1, nn2_w2, nn2_b2, buf1);

    node2_kernel<<<(N_NODES + 255) / 256, 256, 0, stream>>>(
        buf0, buf1, root2, bias2, fc2_w, fc2_b, (float*)d_out);
}
